// SwinSparseAttention3dGroup_41128606826832
// MI455X (gfx1250) — matrix operand order repaired and checked by reading, not compiled, not run on hardware
//
#include <hip/hip_runtime.h>
#include <hip/hip_bf16.h>

// ---------------------------------------------------------------------------
// SwinSparseAttention3d block for MI455X (gfx1250), wave32 + WMMA f16.
// M=100000 voxels, N=8192 queries, K=32 neighbors, C=256, H=8, DH=32, FF=512
//
// GEMMs use block-level macro-tiles: 8 waves cover 128 rows x 64 cols.
// Packed weight fragments for the 4 column-tiles are staged into LDS once per
// block via gfx1250 async LDS copies (global_load_async_to_lds_b128 +
// s_wait_asynccnt), then each wave reads B fragments from LDS and issues
// 4 (or 8, for fused K/V) back-to-back v_wmma_f32_16x16x32_f16 per k-step.
// ---------------------------------------------------------------------------

#define MV   100000
#define NQ   8192
#define KN   32
#define CC   256
#define HH   8
#define DH   32
#define FF   512
#define RX   15
#define RY   15
#define RZ   7

typedef _Float16 v16h __attribute__((ext_vector_type(16)));
typedef _Float16 v8h  __attribute__((ext_vector_type(8)));
typedef float    v8f  __attribute__((ext_vector_type(8)));

__device__ __forceinline__ v8f wmma_f16(v16h a, v16h b, v8f c) {
    return __builtin_amdgcn_wmma_f32_16x16x32_f16(
        /*neg_a=*/false, a, /*neg_b=*/false, b,
        /*c_mod=*/(short)0, c, /*reuse_a=*/false, /*reuse_b=*/false);
}

// A-fragment (16x32 f16, ISA 7.12.2): lane m = lane&15, khi = lane>>4.
// element i<8  -> A[m][k0 + 8*khi + i]        (contiguous 16B)
// element i>=8 -> A[m][k0 + 16 + 8*khi + i-8] (contiguous 16B)
__device__ __forceinline__ v16h load_a_frag(const _Float16* __restrict__ rowptr,
                                            int k0, int khi) {
    v8h lo = *(const v8h*)(rowptr + k0 + 8 * khi);
    v8h hi = *(const v8h*)(rowptr + k0 + 16 + 8 * khi);
    return __builtin_shufflevector(lo, hi, 0,1,2,3,4,5,6,7,8,9,10,11,12,13,14,15);
}

// Async L2 -> LDS copy, 16 bytes per lane (gfx1250 ASYNCcnt path).
__device__ __forceinline__ void async_copy_b128(unsigned lds_off, const void* gaddr) {
    asm volatile("global_load_async_to_lds_b128 %0, %1, off"
                 :: "v"(lds_off), "v"(gaddr) : "memory");
}
__device__ __forceinline__ void wait_async0() {
    asm volatile("s_wait_asynccnt 0x0" ::: "memory");
}

// ---------------------------------------------------------------------------
// Weight pack: W[KK][NN] f32 row-major -> B-fragment order f16.
// B-fragment (KKx16 per tile): lane col n = lane&15, element i -> k = 8*(lane>>4) + (i&7) + 16*(i>>3)
// Packed index: (((ct*(KK/32) + ks)*32 + lane)*16 + i)
// ---------------------------------------------------------------------------
__global__ void __launch_bounds__(256)
pack_w_kernel(const float* __restrict__ W, _Float16* __restrict__ P, int KK, int NN) {
    int tid = blockIdx.x * 256 + threadIdx.x;
    if (tid >= KK * NN) return;
    int i      = tid & 15;
    int lane   = (tid >> 4) & 31;
    int rest   = tid >> 9;                 // ct*(KK/32) + ks
    int kch    = KK >> 5;
    int ks     = rest % kch;
    int ct     = rest / kch;
    int k      = ks * 32 + ((lane >> 4) << 3) + (i & 7) + ((i >> 3) << 4);
    int col    = ct * 16 + (lane & 15);
    P[tid] = (_Float16)W[(size_t)k * NN + col];
}

// ---------------------------------------------------------------------------
// LayerNorm over rows of C=256, f32 in -> f16 out. One block (256 thr) per row.
// ---------------------------------------------------------------------------
__global__ void __launch_bounds__(256)
ln_kernel(const float* __restrict__ in, const float* __restrict__ g,
          const float* __restrict__ b, _Float16* __restrict__ out) {
    __shared__ float red[256];
    const int row = blockIdx.x;
    const int c   = threadIdx.x;
    float x = in[(size_t)row * CC + c];
    red[c] = x;
    __syncthreads();
    for (int s = 128; s > 0; s >>= 1) { if (c < s) red[c] += red[c + s]; __syncthreads(); }
    float mean = red[0] * (1.0f / CC);
    __syncthreads();
    float xc = x - mean;
    red[c] = xc * xc;
    __syncthreads();
    for (int s = 128; s > 0; s >>= 1) { if (c < s) red[c] += red[c + s]; __syncthreads(); }
    float var = red[0] * (1.0f / CC);
    float y = xc * rsqrtf(var + 1e-5f) * g[c] + b[c];
    out[(size_t)row * CC + c] = (_Float16)y;
}

// ---------------------------------------------------------------------------
// q_in = x_h[query_indices] + relu(coords @ w_pos + b_pos). Grid NQ x 256.
// ---------------------------------------------------------------------------
__global__ void __launch_bounds__(256)
qin_kernel(const _Float16* __restrict__ xh, const int* __restrict__ qidx,
           const float* __restrict__ coords, const float* __restrict__ wpos,
           const float* __restrict__ bpos, _Float16* __restrict__ out) {
    const int n = blockIdx.x, c = threadIdx.x;
    float p = bpos[c]
            + coords[n * 3 + 0] * wpos[0 * CC + c]
            + coords[n * 3 + 1] * wpos[1 * CC + c]
            + coords[n * 3 + 2] * wpos[2 * CC + c];
    p = fmaxf(p, 0.0f);
    float v = (float)xh[(size_t)qidx[n] * CC + c] + p;
    out[(size_t)n * CC + c] = (_Float16)v;
}

// ---------------------------------------------------------------------------
// Generic WMMA GEMM with LDS-staged B: Out[R][NN] = A[R][KK] @ Bpacked + bias.
// Block macro-tile 128x64: 8 waves x one 16-row tile each, 4 col tiles shared
// via LDS. MODE 0: f16; 1: relu+f16; 2: f32 (+optional residual).
// Grid: (R/128) * (NN/64) blocks of 256 threads.
// ---------------------------------------------------------------------------
template<int KK, int MODE>
__global__ void __launch_bounds__(256)
gemm_f16_lds_kernel(const _Float16* __restrict__ A, const _Float16* __restrict__ Bp,
                    const float* __restrict__ bias, int R, int NN,
                    void* __restrict__ Out, const float* __restrict__ res) {
    constexpr int KCH = KK >> 5;                 // k-chunks of 32
    __shared__ _Float16 bsh[4 * KCH * 512];      // 4 col tiles of packed B

    const int cblocks = NN >> 6;
    const int rb = blockIdx.x / cblocks;
    const int cb = blockIdx.x % cblocks;

    // ---- async stage of packed B (KCH * 4096 bytes) into LDS ----
    {
        const char* src = (const char*)(Bp + (size_t)cb * 4 * KCH * 512);
        unsigned lbase = (unsigned)(size_t)(&bsh[0]) + threadIdx.x * 16;
        const char* g = src + threadIdx.x * 16;
#pragma unroll
        for (int it = 0; it < KCH; ++it)
            async_copy_b128(lbase + it * 4096, g + it * 4096);
        wait_async0();
    }
    __syncthreads();

    const int lane = threadIdx.x & 31;
    const int wv   = threadIdx.x >> 5;
    const int m = lane & 15, khi = lane >> 4;

    const _Float16* __restrict__ arow = A + (size_t)(rb * 128 + wv * 16 + m) * KK;

    v8f acc[4] = {};
#pragma unroll
    for (int ks = 0; ks < KCH; ++ks) {
        v16h af = load_a_frag(arow, ks * 32, khi);
#pragma unroll
        for (int ct = 0; ct < 4; ++ct) {
            v16h bf = *(const v16h*)(&bsh[((ct * KCH + ks) * 32 + lane) * 16]);
            acc[ct] = wmma_f16(af, bf, acc[ct]);
        }
    }

#pragma unroll
    for (int ct = 0; ct < 4; ++ct) {
        const int col = cb * 64 + ct * 16 + m;
        const float bb = bias ? bias[col] : 0.0f;
#pragma unroll
        for (int j = 0; j < 8; ++j) {
            int row = rb * 128 + wv * 16 + j + 8 * khi;
            float v = acc[ct][j] + bb;
            if (MODE == 1) v = fmaxf(v, 0.0f);
            if (MODE == 2) {
                float o = v + (res ? res[(size_t)row * NN + col] : 0.0f);
                ((float*)Out)[(size_t)row * NN + col] = o;
            } else {
                ((_Float16*)Out)[(size_t)row * NN + col] = (_Float16)v;
            }
        }
    }
}

// ---------------------------------------------------------------------------
// Fused gather + dual GEMM for K and V with LDS-staged weights:
//   Kout/Vout[N*K][C] = mask? bias : x_h[gather_indices] @ Wk/Wv + bias
// Block macro-tile 128x64; A fragment gathered per-lane, shared by 8 WMMAs
// per k-step (4 col tiles x {K,V}). Grid: (N*K/128) * (C/64) blocks.
// ---------------------------------------------------------------------------
__global__ void __launch_bounds__(256)
kv_gemm_lds_kernel(const _Float16* __restrict__ xh,
                   const int* __restrict__ gidx, const unsigned char* __restrict__ gmask,
                   const _Float16* __restrict__ Bpk, const _Float16* __restrict__ Bpv,
                   const float* __restrict__ bk, const float* __restrict__ bv,
                   _Float16* __restrict__ Kout, _Float16* __restrict__ Vout) {
    constexpr int KCH = CC >> 5;                  // 8
    __shared__ _Float16 bshk[4 * KCH * 512];      // 32KB
    __shared__ _Float16 bshv[4 * KCH * 512];      // 32KB

    const int cblocks = CC >> 6;                  // 4
    const int rb = blockIdx.x / cblocks;
    const int cb = blockIdx.x % cblocks;

    // ---- async stage of both packed weight blocks into LDS ----
    {
        const char* sk = (const char*)(Bpk + (size_t)cb * 4 * KCH * 512) + threadIdx.x * 16;
        const char* sv = (const char*)(Bpv + (size_t)cb * 4 * KCH * 512) + threadIdx.x * 16;
        unsigned lk = (unsigned)(size_t)(&bshk[0]) + threadIdx.x * 16;
        unsigned lv = (unsigned)(size_t)(&bshv[0]) + threadIdx.x * 16;
#pragma unroll
        for (int it = 0; it < KCH; ++it) {
            async_copy_b128(lk + it * 4096, sk + it * 4096);
            async_copy_b128(lv + it * 4096, sv + it * 4096);
        }
        wait_async0();
    }
    __syncthreads();

    const int lane = threadIdx.x & 31;
    const int wv   = threadIdx.x >> 5;
    const int m = lane & 15, khi = lane >> 4;

    const int r    = rb * 128 + wv * 16 + m;      // flat (n*K + k)
    const int idx  = gidx[r];
    const bool msk = gmask[r] != 0;
    const _Float16* __restrict__ arow = xh + (size_t)idx * CC;
    if (!msk) __builtin_prefetch(arow, 0, 1);     // global_prefetch_b8

    v8f acck[4] = {}, accv[4] = {};
#pragma unroll
    for (int ks = 0; ks < KCH; ++ks) {
        v16h af = {};
        if (!msk) af = load_a_frag(arow, ks * 32, khi);
#pragma unroll
        for (int ct = 0; ct < 4; ++ct) {
            v16h bfk = *(const v16h*)(&bshk[((ct * KCH + ks) * 32 + lane) * 16]);
            v16h bfv = *(const v16h*)(&bshv[((ct * KCH + ks) * 32 + lane) * 16]);
            acck[ct] = wmma_f16(af, bfk, acck[ct]);
            accv[ct] = wmma_f16(af, bfv, accv[ct]);
        }
    }

#pragma unroll
    for (int ct = 0; ct < 4; ++ct) {
        const int col = cb * 64 + ct * 16 + m;
        const float bbk = bk[col], bbv = bv[col];
#pragma unroll
        for (int j = 0; j < 8; ++j) {
            size_t o = (size_t)(rb * 128 + wv * 16 + j + 8 * khi) * CC + col;
            Kout[o] = (_Float16)(acck[ct][j] + bbk);
            Vout[o] = (_Float16)(accv[ct][j] + bbv);
        }
    }
}

// ---------------------------------------------------------------------------
// Attention core: one wave32 per (n,h). Phase 1: lane = key index (K=32),
// logits + per-axis positional biases + softmax (shfl_xor over 32 lanes).
// Phase 2 (via LDS): lane = channel d, out = sum_k attn[k] * V[n,k,h,d].
// ---------------------------------------------------------------------------
__global__ void __launch_bounds__(256)
attn_kernel(const _Float16* __restrict__ Qh, const _Float16* __restrict__ Kh,
            const _Float16* __restrict__ Vh,
            const unsigned char* __restrict__ gmask,
            const int* __restrict__ relx, const int* __restrict__ rely,
            const int* __restrict__ relz,
            const float* __restrict__ pqx, const float* __restrict__ pqy,
            const float* __restrict__ pqz,
            const float* __restrict__ pkx, const float* __restrict__ pky,
            const float* __restrict__ pkz,
            _Float16* __restrict__ outh) {
    __shared__ float attn_s[8][32];
    const int lane = threadIdx.x & 31;
    const int wv   = threadIdx.x >> 5;
    const int gw   = blockIdx.x * 8 + wv;   // (n,h) pair
    const int n = gw / HH, h = gw % HH;

    const _Float16* __restrict__ qp = Qh + (size_t)n * CC + h * DH;
    const _Float16* __restrict__ kp = Kh + ((size_t)n * KN + lane) * CC + h * DH;

    float q[DH], kk[DH];
#pragma unroll
    for (int d = 0; d < DH; ++d) { q[d] = (float)qp[d]; kk[d] = (float)kp[d]; }

    const int fl = n * KN + lane;
    const int rx = relx[fl], ry = rely[fl], rz = relz[fl];
    const bool msk = gmask[fl] != 0;

    float lg = 0.0f, bias = 0.0f;
#pragma unroll
    for (int d = 0; d < DH; ++d) {
        lg += q[d] * kk[d];
        const int bi = h * DH + d;
        bias += q[d]  * (pqx[bi * RX + rx] + pqy[bi * RY + ry] + pqz[bi * RZ + rz]);
        bias += kk[d] * (pkx[bi * RX + rx] + pky[bi * RY + ry] + pkz[bi * RZ + rz]);
    }
    lg = lg * 0.17677669529663687f + bias;   // DH^-0.5 = 1/sqrt(32)
    if (msk) lg = -1e9f;

    float mx = lg;
#pragma unroll
    for (int off = 16; off > 0; off >>= 1) mx = fmaxf(mx, __shfl_xor(mx, off, 32));
    float e = __expf(lg - mx);
    float s = e;
#pragma unroll
    for (int off = 16; off > 0; off >>= 1) s += __shfl_xor(s, off, 32);

    attn_s[wv][lane] = e / s;
    __syncthreads();

    // Phase 2: lane = channel d within this head
    float acc = 0.0f;
    const _Float16* __restrict__ vbase = Vh + (size_t)n * KN * CC + h * DH + lane;
#pragma unroll 8
    for (int kx = 0; kx < KN; ++kx)
        acc += attn_s[wv][kx] * (float)vbase[(size_t)kx * CC];
    outh[(size_t)n * CC + h * DH + lane] = (_Float16)acc;
}

// ---------------------------------------------------------------------------
// act = voxel_features[query_indices] + proj_out
// ---------------------------------------------------------------------------
__global__ void __launch_bounds__(256)
residual_kernel(const float* __restrict__ voxel, const int* __restrict__ qidx,
                const float* __restrict__ proj, float* __restrict__ act) {
    const int n = blockIdx.x, c = threadIdx.x;
    act[(size_t)n * CC + c] =
        voxel[(size_t)qidx[n] * CC + c] + proj[(size_t)n * CC + c];
}

// ---------------------------------------------------------------------------
extern "C" void kernel_launch(void* const* d_in, const int* in_sizes, int n_in,
                              void* d_out, int out_size, void* d_ws, size_t ws_size,
                              hipStream_t stream) {
    (void)in_sizes; (void)n_in; (void)out_size; (void)ws_size;
    const float* voxel   = (const float*)d_in[0];
    const float* coords  = (const float*)d_in[1];
    const int*   qidx    = (const int*)d_in[2];
    const int*   gidx    = (const int*)d_in[3];
    const unsigned char* gmask = (const unsigned char*)d_in[4];
    const int*   relx    = (const int*)d_in[5];
    const int*   rely    = (const int*)d_in[6];
    const int*   relz    = (const int*)d_in[7];
    const float* w_pos   = (const float*)d_in[8];
    const float* b_pos   = (const float*)d_in[9];
    const float* w_q     = (const float*)d_in[10];
    const float* b_q     = (const float*)d_in[11];
    const float* w_k     = (const float*)d_in[12];
    const float* b_k     = (const float*)d_in[13];
    const float* w_v     = (const float*)d_in[14];
    const float* b_v     = (const float*)d_in[15];
    const float* w_proj  = (const float*)d_in[16];
    const float* b_proj  = (const float*)d_in[17];
    const float* ln1_g   = (const float*)d_in[18];
    const float* ln1_b   = (const float*)d_in[19];
    const float* ln2_g   = (const float*)d_in[20];
    const float* ln2_b   = (const float*)d_in[21];
    const float* w_ff1   = (const float*)d_in[22];
    const float* b_ff1   = (const float*)d_in[23];
    const float* w_ff2   = (const float*)d_in[24];
    const float* b_ff2   = (const float*)d_in[25];
    const float* pqx     = (const float*)d_in[26];
    const float* pqy     = (const float*)d_in[27];
    const float* pqz     = (const float*)d_in[28];
    const float* pkx     = (const float*)d_in[29];
    const float* pky     = (const float*)d_in[30];
    const float* pkz     = (const float*)d_in[31];

    // ---- workspace carve-up (256B aligned) ----
    char* ws = (char*)d_ws;
    size_t off = 0;
    auto alloc = [&](size_t bytes) -> char* {
        char* p = ws + off;
        off = (off + bytes + 255) & ~(size_t)255;
        return p;
    };
    _Float16* xh     = (_Float16*)alloc((size_t)MV * CC * 2);
    _Float16* qin_h  = (_Float16*)alloc((size_t)NQ * CC * 2);
    _Float16* q_h    = (_Float16*)alloc((size_t)NQ * CC * 2);
    _Float16* k_h    = (_Float16*)alloc((size_t)NQ * KN * CC * 2);
    _Float16* v_h    = (_Float16*)alloc((size_t)NQ * KN * CC * 2);
    _Float16* out_h  = (_Float16*)alloc((size_t)NQ * CC * 2);
    float*    projo  = (float*)alloc((size_t)NQ * CC * 4);
    float*    act    = (float*)alloc((size_t)NQ * CC * 4);
    _Float16* ln2h   = (_Float16*)alloc((size_t)NQ * CC * 2);
    _Float16* ff1h   = (_Float16*)alloc((size_t)NQ * FF * 2);
    _Float16* wq_p   = (_Float16*)alloc((size_t)CC * CC * 2);
    _Float16* wk_p   = (_Float16*)alloc((size_t)CC * CC * 2);
    _Float16* wv_p   = (_Float16*)alloc((size_t)CC * CC * 2);
    _Float16* wpr_p  = (_Float16*)alloc((size_t)CC * CC * 2);
    _Float16* wf1_p  = (_Float16*)alloc((size_t)CC * FF * 2);
    _Float16* wf2_p  = (_Float16*)alloc((size_t)FF * CC * 2);

    // ---- 1) weight packing ----
    {
        int e1 = CC * CC, e2 = CC * FF;
        pack_w_kernel<<<(e1 + 255) / 256, 256, 0, stream>>>(w_q,    wq_p,  CC, CC);
        pack_w_kernel<<<(e1 + 255) / 256, 256, 0, stream>>>(w_k,    wk_p,  CC, CC);
        pack_w_kernel<<<(e1 + 255) / 256, 256, 0, stream>>>(w_v,    wv_p,  CC, CC);
        pack_w_kernel<<<(e1 + 255) / 256, 256, 0, stream>>>(w_proj, wpr_p, CC, CC);
        pack_w_kernel<<<(e2 + 255) / 256, 256, 0, stream>>>(w_ff1,  wf1_p, CC, FF);
        pack_w_kernel<<<(e2 + 255) / 256, 256, 0, stream>>>(w_ff2,  wf2_p, FF, CC);
    }

    // ---- 2) LN1 over all voxels -> f16 table (fits in 192MB L2) ----
    ln_kernel<<<MV, 256, 0, stream>>>(voxel, ln1_g, ln1_b, xh);

    // ---- 3) query input assembly ----
    qin_kernel<<<NQ, 256, 0, stream>>>(xh, qidx, coords, w_pos, b_pos, qin_h);

    // ---- 4) Q projection ----
    gemm_f16_lds_kernel<CC, 0><<<(NQ / 128) * (CC / 64), 256, 0, stream>>>(
        qin_h, wq_p, b_q, NQ, CC, (void*)q_h, nullptr);

    // ---- 5) fused gather + K/V projections ----
    kv_gemm_lds_kernel<<<((NQ * KN) / 128) * (CC / 64), 256, 0, stream>>>(
        xh, gidx, gmask, wk_p, wv_p, b_k, b_v, k_h, v_h);

    // ---- 6) attention core ----
    attn_kernel<<<(NQ * HH) / 8, 256, 0, stream>>>(
        q_h, k_h, v_h, gmask, relx, rely, relz,
        pqx, pqy, pqz, pkx, pky, pkz, out_h);

    // ---- 7) output projection (f32) ----
    gemm_f16_lds_kernel<CC, 2><<<(NQ / 128) * (CC / 64), 256, 0, stream>>>(
        out_h, wpr_p, b_proj, NQ, CC, (void*)projo, nullptr);

    // ---- 8) residual with raw voxel features ----
    residual_kernel<<<NQ, 256, 0, stream>>>(voxel, qidx, projo, act);

    // ---- 9) LN2 ----
    ln_kernel<<<NQ, 256, 0, stream>>>(act, ln2_g, ln2_b, ln2h);

    // ---- 10) FFN layer 1 (ReLU, f16 out) ----
    gemm_f16_lds_kernel<CC, 1><<<(NQ / 128) * (FF / 64), 256, 0, stream>>>(
        ln2h, wf1_p, b_ff1, NQ, FF, (void*)ff1h, nullptr);

    // ---- 11) FFN layer 2 (f32 out + residual act) -> d_out ----
    gemm_f16_lds_kernel<FF, 2><<<(NQ / 128) * (CC / 64), 256, 0, stream>>>(
        ff1h, wf2_p, b_ff2, NQ, CC, d_out, act);
}
